// EMB_84842783965559
// MI455X (gfx1250) — matrix operand order repaired and checked by reading, not compiled, not run on hardware
//
#include <hip/hip_runtime.h>
#include <hip/hip_bf16.h>
#include <math.h>

// ---------------------------------------------------------------------------
// MI455X (gfx1250) fused implementation:
//   K1 pack_b : f32 weights -> bf16 WMMA-B-fragment-packed (zero padded)
//   K2 parnet : per-16-row tile: L0(+ELU) -> L1 -> L2 -> constrain ->
//               deterministic per-block column sums (WMMA bf16, f32 acc)
//   K3 preles : fixed-order reduction -> pm; serial 50000-step recurrence
//   K4 resnet : per-16-row tile: L0(+ELU) -> L1 -> L2 -> y_hat
// ---------------------------------------------------------------------------

typedef __bf16 bf16_t;
typedef __attribute__((ext_vector_type(16))) __bf16 v16bf;
typedef __attribute__((ext_vector_type(8)))  __bf16 v8bf;
typedef __attribute__((ext_vector_type(8)))  float  v8f;

#define T_DAYS   50000
#define NBLK     (T_DAYS / 16)   // 3125 row-tiles
#define HDIM     512
#define NT_H     (HDIM / 16)     // 32 N tiles
#define KT_H     (HDIM / 32)     // 16 K tiles
#define LDSTR    520             // bf16 elems per LDS row (1040B: 16B-aligned, bank-safe)
#define INSTR    40              // padded input tile row stride (80B, 16B-aligned)

static constexpr float INF_ = 3.402823466e+38f;

__device__ __constant__ float kScales[27] = {
  400,1,1,1,1,12,10,10,1,1,1,1,1,10,1,1,1,5,1,1,4,1,1,180,1,1,10};
__device__ __constant__ float kLo[27] = {
  0.f,0.f,0.f,-INF_,0.f,0.01f,-0.2f,0.23f,-1.f,0.f,
  0.f,-INF_,-INF_,0.f,0.f,0.f,0.f,0.f,-INF_,0.f,
  0.f,0.f,-INF_,0.f,0.f,0.f,0.f};
__device__ __constant__ float kHi[27] = {
  2.f,INF_,INF_,INF_,2.5f,3.f,2.1f,3.f,0.f,0.7f,
  0.f,INF_,INF_,1.f,1.2f,2.5f,0.f,1.f,INF_,1.3333334f,
  INF_,INF_,INF_,INF_,INF_,INF_,INF_};

// Branchless ELU: for v>0 the exp term is exp(0)-1 = 0 so fmax picks v;
// for v<=0, expm1(v) >= v, so fmax picks the exp branch.  v_exp_f32 path.
__device__ inline float elu_fast(float v) {
  return fmaxf(v, __expf(fminf(v, 0.f)) - 1.f);
}

__device__ inline float constrain_col(float v, int c) {
  if (c == 10 || c == 16) return 1.f / (1.f + __expf(-v));    // sigmoid cols
  return fminf(fmaxf(v, kLo[c]), kHi[c]);                     // clip/relu/id
}

__device__ inline v8f wmma_bf16(v16bf a, v16bf b, v8f c) {
  return __builtin_amdgcn_wmma_f32_16x16x32_bf16(
      /*neg_a=*/false, a, /*neg_b=*/false, b,
      /*c_mod=*/(short)0, c, /*reuse_a=*/false, /*reuse_b=*/false);
}

// A fragment (16x32 bf16, row-major in LDS).  ISA 7.12.2 layout:
// lane-group g = lane>>4, row m = lane&15; elems 0..7 -> K=g*8+j,
// elems 8..15 -> K=16+g*8+j.  Two aligned 16-byte LDS vector loads.
__device__ inline v16bf load_a_lds(const bf16_t* base, int stride, int kt, int lane) {
  const int m = lane & 15, g = lane >> 4;
  const bf16_t* row = base + m * stride + kt * 32 + g * 8;
  const v8bf lo = *(const v8bf*)(row);
  const v8bf hi = *(const v8bf*)(row + 16);
  return __builtin_shufflevector(lo, hi, 0,1,2,3,4,5,6,7,8,9,10,11,12,13,14,15);
}

// B fragment: one aligned 32-byte load per lane from the packed buffer.
__device__ inline v16bf load_b_pk(const bf16_t* pk, int ntiles, int kt, int nt, int lane) {
  return *(const v16bf*)(pk + ((size_t)((kt * ntiles + nt) * 32 + lane)) * 16);
}

// ---------------------------------------------------------------------------
// K1: pack W[K][N] (f32, row-major) into bf16 B-fragment order, zero padded.
// out[((kt*Ntiles+nt)*32+lane)*16+j] = W[kt*32+(lane>>4)*8+(j&7)+(j>>3)*16][nt*16+(lane&15)]
// ---------------------------------------------------------------------------
__global__ void pack_b_kernel(const float* __restrict__ W, int K, int N,
                              int Ktiles, int Ntiles, bf16_t* __restrict__ out) {
  const int total = Ktiles * Ntiles * 512;
  for (int idx = blockIdx.x * blockDim.x + threadIdx.x; idx < total;
       idx += gridDim.x * blockDim.x) {
    const int j    = idx & 15;
    const int lane = (idx >> 4) & 31;
    const int tile = idx >> 9;
    const int nt   = tile % Ntiles;
    const int kt   = tile / Ntiles;
    const int k = kt * 32 + (lane >> 4) * 8 + (j & 7) + ((j >> 3) << 4);
    const int n = nt * 16 + (lane & 15);
    const float v = (k < K && n < N) ? W[k * N + n] : 0.f;
    out[idx] = (bf16_t)v;
  }
}

// ---------------------------------------------------------------------------
// K2: fused parnet for one 16-row tile of T.
// ---------------------------------------------------------------------------
__global__ __launch_bounds__(256)
void parnet_kernel(const float* __restrict__ x,
                   const bf16_t* __restrict__ w0p, const float* __restrict__ b0,
                   const bf16_t* __restrict__ w1p, const float* __restrict__ b1,
                   const bf16_t* __restrict__ w2p, const float* __restrict__ b2,
                   float* __restrict__ partial) {
  __shared__ __align__(16) bf16_t sIn[16 * INSTR];
  __shared__ __align__(16) bf16_t sH0[16 * LDSTR];
  __shared__ __align__(16) bf16_t sH1[16 * LDSTR];
  const int tid = threadIdx.x, lane = tid & 31, w = tid >> 5;
  const int t0 = blockIdx.x * 16;

  // stage x tile (16x12 f32 -> 16x32 bf16, zero padded)
  for (int e = tid; e < 16 * 32; e += 256) {
    const int r = e >> 5, k = e & 31;
    const float v = (k < 12) ? x[(t0 + r) * 12 + k] : 0.f;
    sIn[r * INSTR + k] = (bf16_t)v;
  }
  __syncthreads();

  // ---- layer 0: h0 = ELU(x @ W0 + b0), K=32 (one WMMA per tile) ----
  {
    const v16bf a = load_a_lds(sIn, INSTR, 0, lane);
#pragma unroll
    for (int i = 0; i < 4; ++i) {
      const int nt = w + 8 * i;
      v8f c = {};
      const v16bf b = load_b_pk(w0p, NT_H, 0, nt, lane);
      c = wmma_bf16(a, b, c);
      const int col = nt * 16 + (lane & 15);
      const float bv = b0[col];
#pragma unroll
      for (int r = 0; r < 8; ++r)
        sH0[((lane >> 4) * 8 + r) * LDSTR + col] = (bf16_t)elu_fast(c[r] + bv);
    }
  }
  __syncthreads();

  // ---- layer 1: h1 = h0 @ W1 + b1 (no activation) ----
  {
    v8f acc[4] = {{}, {}, {}, {}};
#pragma unroll 4
    for (int kt = 0; kt < KT_H; ++kt) {
      const v16bf a = load_a_lds(sH0, LDSTR, kt, lane);
#pragma unroll
      for (int i = 0; i < 4; ++i) {
        const v16bf b = load_b_pk(w1p, NT_H, kt, w + 8 * i, lane);
        acc[i] = wmma_bf16(a, b, acc[i]);
      }
    }
#pragma unroll
    for (int i = 0; i < 4; ++i) {
      const int col = (w + 8 * i) * 16 + (lane & 15);
      const float bv = b1[col];
#pragma unroll
      for (int r = 0; r < 8; ++r)
        sH1[((lane >> 4) * 8 + r) * LDSTR + col] = (bf16_t)(acc[i][r] + bv);
    }
  }
  __syncthreads();

  // ---- layer 2: praw = h1 @ W2 + b2 (27 cols, padded 32), constrain, sum ----
  if (w < 2) {
    const int nt = w;
    v8f c = {};
#pragma unroll 4
    for (int kt = 0; kt < KT_H; ++kt) {
      const v16bf a = load_a_lds(sH1, LDSTR, kt, lane);
      const v16bf b = load_b_pk(w2p, 2, kt, nt, lane);
      c = wmma_bf16(a, b, c);
    }
    const int col = nt * 16 + (lane & 15);
    const float bv = (col < 27) ? b2[col] : 0.f;
    float part = 0.f;
#pragma unroll
    for (int r = 0; r < 8; ++r) {
      float v = c[r] + bv;
      if (col < 27) v = constrain_col(v, col);
      part += v;
    }
    part += __shfl_xor(part, 16, 32);   // fold rows 8..15 into 0..7 half
    if (lane < 16 && col < 27)
      partial[blockIdx.x * 32 + col] = part;   // deterministic (no atomics)
  }
}

// ---------------------------------------------------------------------------
// K3: reduce column partials -> pm, then serial PRELES scan (lane 0).
// ---------------------------------------------------------------------------
__global__ void preles_kernel(const float* __restrict__ partial,
                              const float* __restrict__ cin,
                              const float* __restrict__ sw,
                              float* __restrict__ out) {
  __shared__ float pm[27];
  const int l = threadIdx.x;
  if (l < 27) {
    float s = 0.f;
    for (int b = 0; b < NBLK; ++b) s += partial[b * 32 + l];
    pm[l] = s * (1.f / (float)T_DAYS) * kScales[l];
  }
  __syncthreads();
  if (l != 0) return;

  const float soildepth = pm[0],  ThetaFC = pm[1],  ThetaPWP = pm[2];
  const float tauDrainage = pm[3], beta = pm[4],    tau = pm[5];
  const float S0 = pm[6], Smax = pm[7], kappa = pm[8], gamma = pm[9];
  const float soilthres = pm[10], bCO2 = pm[11], xCO2 = pm[12];
  const float ETbeta = pm[13], ETkappa = pm[14], ETchi = pm[15];
  const float ETsoilthres = pm[16], ETnu = pm[17], MeltCoef = pm[18];
  const float I0 = pm[19], CWmax = pm[20], SnowThreshold = pm[21];
  const float Tzero = pm[22], SWinit = pm[23], CWinit = pm[24];
  const float SOGinit = pm[25], Sinit = pm[26];
  const float fcap = ThetaFC * soildepth;
  const float sw0 = sw[0], sw1 = sw[1];

  float S = Sinit, theta = SWinit, snow = SOGinit, canw = CWinit;
  for (int t = 0; t < T_DAYS; ++t) {
    const float* d = cin + t * 7;
    const float precip = d[0], tair = d[1], par = d[2];
    const float vpd = d[3], fapar = d[4], co2 = d[6];

    S += (tair - S) / tau;
    const float fS = fminf(fmaxf(S - S0, 0.f) / Smax, 1.f);
    const float thetavol = theta / soildepth;
    const float REW = (thetavol - ThetaPWP) / (ThetaFC - ThetaPWP);
    const float fD = fminf(__expf(kappa * vpd), 1.f);
    const float fW = (REW < soilthres) ? ((REW > 0.01f) ? REW / soilthres : 0.f) : 1.f;
    const float fL = 1.f / (gamma * par + 1.f);
    const float fE = fminf(fD, fW);
    const float gpp380 = beta * par * fapar * fS * fL * fE;
    const float logco2 = __logf(co2 / 380.f);
    const float gpp = gpp380 * (1.f + bCO2 * logco2);

    const float intercepted = (tair > SnowThreshold) ? precip * I0 * fapar / 0.75f : 0.f;
    const float rain = precip - intercepted;
    const float cap = CWmax * fapar;
    const bool over = (intercepted + canw > cap);
    const bool small_cw = (CWmax <= 1e-8f);
    float tf = small_cw ? (rain + intercepted)
                        : (over ? rain + intercepted + canw - cap : rain);
    canw = small_cw ? canw : (over ? cap : canw + intercepted);

    const float newsnow = (tair < SnowThreshold) ? tf : 0.f;
    tf -= newsnow;
    float melt = (tair >= Tzero) ? MeltCoef * (tair - Tzero) : 0.f;
    melt = fminf(melt, snow + newsnow);
    snow = snow + newsnow - melt;

    float fWet = (REW < ETsoilthres) ? ((REW > 0.01f) ? REW / ETsoilthres : 0.f) : 1.f;
    fWet = (canw > 1e-8f) ? 1.f : fWet;
    const float fCO2et = 1.f + xCO2 * logco2;
    const float transp = vpd * ETbeta * gpp380 / __powf(vpd, ETkappa) *
                         __powf(fmaxf(fW, 1e-12f), ETnu) * fCO2et;
    const float evap = ETchi * (1.f - fapar) * fWet * par;
    const float et = transp + evap;

    const float canw2 = fmaxf(canw - et, 0.f);
    const float rem = fmaxf(et - canw, 0.f);
    const float snow2 = fmaxf(snow - rem, 0.f);
    const float etsoil = fmaxf(rem - snow, 0.f);
    const float st0 = fmaxf(theta + tf + melt - etsoil, 1e-4f);
    const float drain = (tauDrainage > 0.f) ? fmaxf(st0 - fcap, 0.f) / tauDrainage : 0.f;
    theta = st0 - drain;
    canw = canw2;
    snow = snow2;

    float* pp = out + 2 * T_DAYS + t * 3;
    pp[0] = gpp;
    pp[1] = et;
    pp[2] = (theta - sw0) / sw1;
  }
}

// ---------------------------------------------------------------------------
// K4: fused resnet for one 16-row tile; writes y_hat [T,2].
// ---------------------------------------------------------------------------
__global__ __launch_bounds__(256)
void resnet_kernel(const float* __restrict__ pp,
                   const bf16_t* __restrict__ w0p, const float* __restrict__ b0,
                   const bf16_t* __restrict__ w1p, const float* __restrict__ b1,
                   const bf16_t* __restrict__ w2p, const float* __restrict__ b2,
                   float* __restrict__ yhat) {
  __shared__ __align__(16) bf16_t sIn[16 * INSTR];
  __shared__ __align__(16) bf16_t sH0[16 * LDSTR];
  __shared__ __align__(16) bf16_t sH1[16 * LDSTR];
  const int tid = threadIdx.x, lane = tid & 31, w = tid >> 5;
  const int t0 = blockIdx.x * 16;

  for (int e = tid; e < 16 * 32; e += 256) {
    const int r = e >> 5, k = e & 31;
    const float v = (k < 3) ? pp[(t0 + r) * 3 + k] : 0.f;
    sIn[r * INSTR + k] = (bf16_t)v;
  }
  __syncthreads();

  // layer 0 + ELU
  {
    const v16bf a = load_a_lds(sIn, INSTR, 0, lane);
#pragma unroll
    for (int i = 0; i < 4; ++i) {
      const int nt = w + 8 * i;
      v8f c = {};
      const v16bf b = load_b_pk(w0p, NT_H, 0, nt, lane);
      c = wmma_bf16(a, b, c);
      const int col = nt * 16 + (lane & 15);
      const float bv = b0[col];
#pragma unroll
      for (int r = 0; r < 8; ++r)
        sH0[((lane >> 4) * 8 + r) * LDSTR + col] = (bf16_t)elu_fast(c[r] + bv);
    }
  }
  __syncthreads();

  // layer 1 (no activation)
  {
    v8f acc[4] = {{}, {}, {}, {}};
#pragma unroll 4
    for (int kt = 0; kt < KT_H; ++kt) {
      const v16bf a = load_a_lds(sH0, LDSTR, kt, lane);
#pragma unroll
      for (int i = 0; i < 4; ++i) {
        const v16bf b = load_b_pk(w1p, NT_H, kt, w + 8 * i, lane);
        acc[i] = wmma_bf16(a, b, acc[i]);
      }
    }
#pragma unroll
    for (int i = 0; i < 4; ++i) {
      const int col = (w + 8 * i) * 16 + (lane & 15);
      const float bv = b1[col];
#pragma unroll
      for (int r = 0; r < 8; ++r)
        sH1[((lane >> 4) * 8 + r) * LDSTR + col] = (bf16_t)(acc[i][r] + bv);
    }
  }
  __syncthreads();

  // layer 2: y = h1 @ W2 + b2 (2 cols, padded 16) -> global
  if (w == 0) {
    v8f c = {};
#pragma unroll 4
    for (int kt = 0; kt < KT_H; ++kt) {
      const v16bf a = load_a_lds(sH1, LDSTR, kt, lane);
      const v16bf b = load_b_pk(w2p, 1, kt, 0, lane);
      c = wmma_bf16(a, b, c);
    }
    const int col = lane & 15;
    if (col < 2) {
      const float bv = b2[col];
      const int rbase = (lane >> 4) * 8;
#pragma unroll
      for (int r = 0; r < 8; ++r)
        yhat[(t0 + rbase + r) * 2 + col] = c[r] + bv;
    }
  }
}

// ---------------------------------------------------------------------------
// Workspace layout (bytes, 256-aligned)
// ---------------------------------------------------------------------------
static constexpr size_t OFF_PART = 0;                       // 3125*32 f32 = 400000
static constexpr size_t OFF_W0P  = 400128;                  // 32x512  bf16 = 32768
static constexpr size_t OFF_W1P  = OFF_W0P + 32768;         // 512x512 bf16 = 524288
static constexpr size_t OFF_W2P  = OFF_W1P + 524288;        // 512x32  bf16 = 32768
static constexpr size_t OFF_U0P  = OFF_W2P + 32768;         // 32x512  bf16 = 32768
static constexpr size_t OFF_U1P  = OFF_U0P + 32768;         // 512x512 bf16 = 524288
static constexpr size_t OFF_U2P  = OFF_U1P + 524288;        // 512x16  bf16 = 16384

extern "C" void kernel_launch(void* const* d_in, const int* in_sizes, int n_in,
                              void* d_out, int out_size, void* d_ws, size_t ws_size,
                              hipStream_t stream) {
  (void)in_sizes; (void)n_in; (void)out_size; (void)ws_size;
  const float* x   = (const float*)d_in[0];
  const float* cin = (const float*)d_in[1];
  const float* sw  = (const float*)d_in[2];
  const float* pW0 = (const float*)d_in[4];
  const float* pb0 = (const float*)d_in[5];
  const float* pW1 = (const float*)d_in[6];
  const float* pb1 = (const float*)d_in[7];
  const float* pW2 = (const float*)d_in[8];
  const float* pb2 = (const float*)d_in[9];
  const float* rW0 = (const float*)d_in[10];
  const float* rb0 = (const float*)d_in[11];
  const float* rW1 = (const float*)d_in[12];
  const float* rb1 = (const float*)d_in[13];
  const float* rW2 = (const float*)d_in[14];
  const float* rb2 = (const float*)d_in[15];
  float* out = (float*)d_out;

  char* ws = (char*)d_ws;
  float*  partial = (float*)(ws + OFF_PART);
  bf16_t* w0p = (bf16_t*)(ws + OFF_W0P);
  bf16_t* w1p = (bf16_t*)(ws + OFF_W1P);
  bf16_t* w2p = (bf16_t*)(ws + OFF_W2P);
  bf16_t* u0p = (bf16_t*)(ws + OFF_U0P);
  bf16_t* u1p = (bf16_t*)(ws + OFF_U1P);
  bf16_t* u2p = (bf16_t*)(ws + OFF_U2P);

  // K1: pack weights (K padded to mult-32 with zeros, N to mult-16)
  pack_b_kernel<<<64,   256, 0, stream>>>(pW0,  12, 512,  1, 32, w0p);
  pack_b_kernel<<<1024, 256, 0, stream>>>(pW1, 512, 512, 16, 32, w1p);
  pack_b_kernel<<<64,   256, 0, stream>>>(pW2, 512,  27, 16,  2, w2p);
  pack_b_kernel<<<64,   256, 0, stream>>>(rW0,   3, 512,  1, 32, u0p);
  pack_b_kernel<<<1024, 256, 0, stream>>>(rW1, 512, 512, 16, 32, u1p);
  pack_b_kernel<<<32,   256, 0, stream>>>(rW2, 512,   2, 16,  1, u2p);

  // K2: fused parnet + constrained column partial sums
  parnet_kernel<<<NBLK, 256, 0, stream>>>(x, w0p, pb0, w1p, pb1, w2p, pb2, partial);

  // K3: deterministic reduction -> pm, serial PRELES scan -> ppreds (out+2T)
  preles_kernel<<<1, 32, 0, stream>>>(partial, cin, sw, out);

  // K4: fused resnet on ppreds -> y_hat (out[0 : 2T))
  resnet_kernel<<<NBLK, 256, 0, stream>>>(out + 2 * T_DAYS,
                                          u0p, rb0, u1p, rb1, u2p, rb2, out);
}
